// RVQ_gcn_47072841564920
// MI455X (gfx1250) — matrix operand order repaired and checked by reading, not compiled, not run on hardware
//
#include <hip/hip_runtime.h>

// ---------------------------------------------------------------------------
// Residual VQ (SoundStream) for MI455X / gfx1250.
//   x:         [16, 2048, 512] f32   -> rows M = 32768, depth D = 512
//   codebooks: [8, 1024, 512]  f32   -> Q = 8 stages, C = 1024 codes
// Outputs (concatenated in d_out, f32):
//   quantized  [16,2048,512] (16777216), indices [8,16,2048] (262144), loss (1)
//
// Distance GEMM per stage uses v_wmma_f32_16x16x32_bf16 (bf16 A/B, f32 acc)
// with double-buffered LDS fed by GLOBAL_LOAD_ASYNC_TO_LDS_B128 (ASYNCcnt);
// residual/loss updates stay in f32 from the original codebooks.
// ---------------------------------------------------------------------------

typedef __attribute__((ext_vector_type(16))) __bf16 v16bf;
typedef __attribute__((ext_vector_type(8)))  __bf16 v8bf;
typedef __attribute__((ext_vector_type(4)))  __bf16 v4bf;
typedef __attribute__((ext_vector_type(8)))  float  v8f;
typedef __attribute__((ext_vector_type(4)))  int    v4i;

// Address-space-qualified pointer types for the async-to-LDS builtin:
// param 0 is v4i addrspace(1)* (global), param 1 is v4i addrspace(3)* (LDS).
typedef __attribute__((address_space(1))) v4i* gas_v4i_ptr;
typedef __attribute__((address_space(3))) v4i* las_v4i_ptr;

#define MT      32768      // rows (B*T)
#define DK      512        // depth
#define CC      1024       // codes per stage
#define QS      8          // stages
#define NELEM   (MT * DK)  // 16777216

#define BLK_M   128
#define BLK_N   128
#define BLK_K   64
#define LDSS    72         // LDS row stride in bf16 (64 + 8 pad, 144B: 16B aligned)
#define NCBLK   (CC / BLK_N)  // 8 column blocks

static __device__ __forceinline__ v16bf pack16(v8bf lo, v8bf hi) {
    union { v16bf v; v8bf h[2]; } u;
    u.h[0] = lo; u.h[1] = hi;
    return u.v;
}

// 16B/lane async copy global -> LDS (ASYNCcnt-tracked, no VGPR round trip).
static __device__ __forceinline__ void async_copy16(__bf16* lds, const __bf16* g) {
#if __has_builtin(__builtin_amdgcn_global_load_async_to_lds_b128)
    __builtin_amdgcn_global_load_async_to_lds_b128(
        (gas_v4i_ptr)g, (las_v4i_ptr)lds, 0, 0);
#else
    const unsigned loff =
        (unsigned)(size_t)(__attribute__((address_space(3))) __bf16*)lds;
    asm volatile("global_load_async_to_lds_b128 %0, %1, off"
                 :: "v"(loff), "v"(g) : "memory");
#endif
}

static __device__ __forceinline__ void wait_async0() {
#if __has_builtin(__builtin_amdgcn_s_wait_asynccnt)
    __builtin_amdgcn_s_wait_asynccnt(0);
#else
    asm volatile("s_wait_asynccnt 0x0" ::: "memory");
#endif
}

// ---------------------------------------------------------------------------
// Stage GEMM + partial argmin.
// Block: 256 threads (8 waves). Tile: 128 rows x 128 cols, K looped by 64,
// double-buffered: async-copy chunk k+1 into buf^1 while computing chunk k.
// Wave w owns rows [w*16, w*16+16) x all 128 cols -> 8 accum tiles (64 VGPRs).
// Fragment loads follow the CDNA5 ISA VGPR layouts:
//   A (16x32 bf16): lanes 0-15 row=lane, K {0..7,16..23}; lanes 16-31 row=lane-16,
//                   K {8..15,24..31}  -> two 16B LDS loads per fragment.
//   B (32x16 bf16): lane column = lane%16; lanes 0-15 K {0..7,16..23}, lanes 16-31
//                   K {8..15,24..31}  -> two 16B LDS loads per fragment.
// ---------------------------------------------------------------------------
__global__ __launch_bounds__(256) void rvq_gemm_argmin(
    const __bf16* __restrict__ Rbf,    // [MT, DK] residual, bf16
    const __bf16* __restrict__ Ebf,    // [CC, DK] stage codebook, bf16
    const float*  __restrict__ cnorm,  // [CC] ||e||^2, f32
    float*        __restrict__ partMin,// [NCBLK][MT]
    int*          __restrict__ partIdx)// [NCBLK][MT]
{
    __shared__ __bf16 As[2][BLK_M * LDSS];
    __shared__ __bf16 Bs[2][BLK_N * LDSS];
    __shared__ float  Cn[BLK_N];

    const int tid  = threadIdx.x;
    const int wave = tid >> 5;
    const int lane = tid & 31;
    const int l16  = lane & 15;
    const int hi   = lane >> 4;   // 0 = lanes 0-15, 1 = lanes 16-31

    const int mBlk  = (int)blockIdx.x / NCBLK;
    const int nBlk  = (int)blockIdx.x % NCBLK;
    const int mBase = mBlk * BLK_M;
    const int cBase = nBlk * BLK_N;

    if (tid < BLK_N) Cn[tid] = cnorm[cBase + tid];

    v8f acc[8];
#pragma unroll
    for (int n = 0; n < 8; ++n)
#pragma unroll
        for (int i = 0; i < 8; ++i) acc[n][i] = 0.0f;

    // Prologue: async-stage chunk 0 into buffer 0.
#pragma unroll
    for (int c = tid; c < (BLK_M * BLK_K / 8); c += 256) {
        const int row = c >> 3;
        const int kc  = (c & 7) << 3;
        async_copy16(&As[0][row * LDSS + kc], Rbf + (size_t)(mBase + row) * DK + kc);
        async_copy16(&Bs[0][row * LDSS + kc], Ebf + (size_t)(cBase + row) * DK + kc);
    }
    wait_async0();
    __syncthreads();

    int buf = 0;
    for (int kb = 0; kb < DK; kb += BLK_K, buf ^= 1) {
        // Issue async copies for the next chunk into the other buffer.
        if (kb + BLK_K < DK) {
#pragma unroll
            for (int c = tid; c < (BLK_M * BLK_K / 8); c += 256) {
                const int row = c >> 3;
                const int kc  = (c & 7) << 3;
                async_copy16(&As[buf ^ 1][row * LDSS + kc],
                             Rbf + (size_t)(mBase + row) * DK + kb + BLK_K + kc);
                async_copy16(&Bs[buf ^ 1][row * LDSS + kc],
                             Ebf + (size_t)(cBase + row) * DK + kb + BLK_K + kc);
            }
        }

        const __bf16* lA = As[buf];
        const __bf16* lB = Bs[buf];
#pragma unroll
        for (int kk = 0; kk < BLK_K; kk += 32) {
            // A fragment for this wave's 16 rows
            const int ar    = wave * 16 + l16;
            const int akoff = kk + hi * 8;
            const v8bf alo = *(const v8bf*)(lA + ar * LDSS + akoff);
            const v8bf ahi = *(const v8bf*)(lA + ar * LDSS + akoff + 16);
            const v16bf aF = pack16(alo, ahi);

#pragma unroll
            for (int n = 0; n < 8; ++n) {
                const int br    = n * 16 + l16;
                const int bkoff = kk + hi * 8;
                const v8bf blo = *(const v8bf*)(lB + br * LDSS + bkoff);
                const v8bf bhi = *(const v8bf*)(lB + br * LDSS + bkoff + 16);
                const v16bf bF = pack16(blo, bhi);
                acc[n] = __builtin_amdgcn_wmma_f32_16x16x32_bf16(
                    false, aF, false, bF, (short)0, acc[n], false, false);
            }
        }

        // All of this wave's async copies into buf^1 must land, then rendezvous.
        wait_async0();
        __syncthreads();
    }

    // Epilogue: score = ||e||^2 - 2*(r.e); per-row argmin over 128 cols.
    // C/D layout: VGPR v, lanes 0-15 -> row v, col lane; lanes 16-31 -> row v+8.
    const int rowBase = mBase + wave * 16;
#pragma unroll
    for (int v = 0; v < 8; ++v) {
        float best = 3.4e38f;
        int   bidx = 0;
#pragma unroll
        for (int n = 0; n < 8; ++n) {
            const float s = Cn[n * 16 + l16] - 2.0f * acc[n][v];
            const int   c = cBase + n * 16 + l16;
            if (s < best) { best = s; bidx = c; }   // n ascending -> first-min kept
        }
        // butterfly across the 16 lanes of this half-wave (rows differ per half)
#pragma unroll
        for (int off = 1; off < 16; off <<= 1) {
            const float ob = __shfl_xor(best, off, 32);
            const int   oi = __shfl_xor(bidx, off, 32);
            if (ob < best || (ob == best && oi < bidx)) { best = ob; bidx = oi; }
        }
        if (l16 == 0) {
            const int grow = rowBase + v + hi * 8;
            partMin[(size_t)nBlk * MT + grow] = best;
            partIdx[(size_t)nBlk * MT + grow] = bidx;
        }
    }
}

// Reduce 8 column-block partials -> final index per row (float copy into d_out).
__global__ __launch_bounds__(256) void rvq_argmin_reduce(
    const float* __restrict__ partMin, const int* __restrict__ partIdx,
    int* __restrict__ idxInt, float* __restrict__ idxF)
{
    const int row = (int)blockIdx.x * 256 + (int)threadIdx.x;
    if (row >= MT) return;
    float best = partMin[row];
    int   bi   = partIdx[row];
#pragma unroll
    for (int b = 1; b < NCBLK; ++b) {
        const float v = partMin[(size_t)b * MT + row];
        const int   i = partIdx[(size_t)b * MT + row];
        if (v < best || (v == best && i < bi)) { best = v; bi = i; }
    }
    idxInt[row] = bi;
    idxF[row]   = (float)bi;
}

// Gather f32 code, update residual in place, emit bf16 residual for the next
// stage's GEMM, and accumulate sum((r - q)^2) into the loss accumulator.
__global__ __launch_bounds__(128) void rvq_update(
    float* __restrict__ residual, __bf16* __restrict__ Rbf,
    const float* __restrict__ cb, const int* __restrict__ idx,
    float* __restrict__ lossAccum)
{
    __shared__ float wsum[4];
    const int row  = (int)blockIdx.x;
    const int t    = (int)threadIdx.x;
    const int code = idx[row];

    const float4 rv = ((const float4*)(residual + (size_t)row * DK))[t];
    const float4 ev = ((const float4*)(cb + (size_t)code * DK))[t];
    float4 d;
    d.x = rv.x - ev.x; d.y = rv.y - ev.y; d.z = rv.z - ev.z; d.w = rv.w - ev.w;
    ((float4*)(residual + (size_t)row * DK))[t] = d;

    v4bf b;
    b[0] = (__bf16)d.x; b[1] = (__bf16)d.y; b[2] = (__bf16)d.z; b[3] = (__bf16)d.w;
    ((v4bf*)(Rbf + (size_t)row * DK))[t] = b;

    float ls = d.x * d.x + d.y * d.y + d.z * d.z + d.w * d.w;
#pragma unroll
    for (int off = 16; off > 0; off >>= 1) ls += __shfl_down(ls, off, 32);
    if ((t & 31) == 0) wsum[t >> 5] = ls;
    __syncthreads();
    if (t == 0) atomicAdd(lossAccum, wsum[0] + wsum[1] + wsum[2] + wsum[3]);
}

// residual := x, bf16 copy, zero loss accumulator.
__global__ __launch_bounds__(256) void rvq_init(
    const float* __restrict__ x, float* __restrict__ residual,
    __bf16* __restrict__ Rbf, float* __restrict__ lossAccum)
{
    const size_t i = (size_t)blockIdx.x * 256 + threadIdx.x;  // x4 floats
    const float4 v = ((const float4*)x)[i];
    ((float4*)residual)[i] = v;
    v4bf b;
    b[0] = (__bf16)v.x; b[1] = (__bf16)v.y; b[2] = (__bf16)v.z; b[3] = (__bf16)v.w;
    ((v4bf*)Rbf)[i] = b;
    if (i == 0) *lossAccum = 0.0f;
}

// bf16 copy of all 8 codebooks.
__global__ __launch_bounds__(256) void rvq_conv_cb(
    const float* __restrict__ cb, __bf16* __restrict__ Ebf)
{
    const size_t i = (size_t)blockIdx.x * 256 + threadIdx.x;  // x4 floats
    const float4 v = ((const float4*)cb)[i];
    v4bf b;
    b[0] = (__bf16)v.x; b[1] = (__bf16)v.y; b[2] = (__bf16)v.z; b[3] = (__bf16)v.w;
    ((v4bf*)Ebf)[i] = b;
}

// ||e||^2 for all Q*C code rows (exact f32).
__global__ __launch_bounds__(256) void rvq_cnorm(
    const float* __restrict__ cb, float* __restrict__ cnorm)
{
    const int r = (int)blockIdx.x * 256 + (int)threadIdx.x;
    if (r >= QS * CC) return;
    const float4* p = (const float4*)(cb + (size_t)r * DK);
    float s = 0.0f;
    for (int j = 0; j < DK / 4; ++j) {
        const float4 v = p[j];
        s += v.x * v.x + v.y * v.y + v.z * v.z + v.w * v.w;
    }
    cnorm[r] = s;
}

// quantized = x - residual_final (in place over the residual buffer in d_out);
// thread 0 writes loss = BETA * sum / N.
__global__ __launch_bounds__(256) void rvq_finalize(
    const float* __restrict__ x, float* __restrict__ outQ,
    const float* __restrict__ lossAccum, float* __restrict__ outLoss)
{
    const size_t i = (size_t)blockIdx.x * 256 + threadIdx.x;  // x4 floats
    const float4 xv = ((const float4*)x)[i];
    const float4 rv = ((float4*)outQ)[i];
    float4 q;
    q.x = xv.x - rv.x; q.y = xv.y - rv.y; q.z = xv.z - rv.z; q.w = xv.w - rv.w;
    ((float4*)outQ)[i] = q;
    if (i == 0) *outLoss = 0.25f * (*lossAccum) * (1.0f / (float)NELEM);
}

// ---------------------------------------------------------------------------
extern "C" void kernel_launch(void* const* d_in, const int* in_sizes, int n_in,
                              void* d_out, int out_size, void* d_ws, size_t ws_size,
                              hipStream_t stream)
{
    const float* x  = (const float*)d_in[0];     // [16,2048,512]
    const float* cb = (const float*)d_in[1];     // [8,1024,512]

    float* outQ    = (float*)d_out;              // residual during stages, quantized at end
    float* outIdxF = (float*)d_out + NELEM;      // [8][32768] indices as float
    float* outLoss = (float*)d_out + NELEM + (size_t)QS * MT;

    char* ws = (char*)d_ws;
    __bf16* Rbf   = (__bf16*)ws;  ws += (size_t)MT * DK * 2;        // 32 MB
    __bf16* Ebf   = (__bf16*)ws;  ws += (size_t)QS * CC * DK * 2;   // 8 MB
    float*  cnorm = (float*)ws;   ws += (size_t)QS * CC * 4;        // 32 KB
    float*  pMin  = (float*)ws;   ws += (size_t)NCBLK * MT * 4;     // 1 MB
    int*    pIdx  = (int*)ws;     ws += (size_t)NCBLK * MT * 4;     // 1 MB
    int*    idxI  = (int*)ws;     ws += (size_t)MT * 4;             // 128 KB
    float*  lossA = (float*)ws;   ws += 256;

    // Prologue
    rvq_init<<<NELEM / 4 / 256, 256, 0, stream>>>(x, outQ, Rbf, lossA);
    rvq_conv_cb<<<(QS * CC * DK) / 4 / 256, 256, 0, stream>>>(cb, Ebf);
    rvq_cnorm<<<(QS * CC + 255) / 256, 256, 0, stream>>>(cb, cnorm);

    // 8 sequential stages
    for (int q = 0; q < QS; ++q) {
        rvq_gemm_argmin<<<(MT / BLK_M) * NCBLK, 256, 0, stream>>>(
            Rbf, Ebf + (size_t)q * CC * DK, cnorm + (size_t)q * CC, pMin, pIdx);
        rvq_argmin_reduce<<<MT / 256, 256, 0, stream>>>(
            pMin, pIdx, idxI, outIdxF + (size_t)q * MT);
        rvq_update<<<MT, 128, 0, stream>>>(
            outQ, Rbf, cb + (size_t)q * CC * DK, idxI, lossA);
    }

    // Epilogue
    rvq_finalize<<<NELEM / 4 / 256, 256, 0, stream>>>(x, outQ, lossA, outLoss);
}
